// SparseTopKDispatcher_73796128080299
// MI455X (gfx1250) — matrix operand order repaired and checked by reading, not compiled, not run on hardware
//
#include <hip/hip_runtime.h>
#include <hip/hip_bf16.h>

// Problem constants (from reference): B=2, S=2048, D=1024, F=4096, E=8, K=2
#define TOK   4096   // B*S tokens
#define DDIM  1024
#define FDIM  4096
#define NEXP  8
#define TOPK  2
#define NPAIR 8192   // TOK*TOPK

typedef __attribute__((ext_vector_type(16))) __bf16 v16bf;
typedef __attribute__((ext_vector_type(8)))  float  v8f;

__device__ __forceinline__ float gelu_fast(float x) {
  // tanh-gelu, branch-free: 0.5x(1+tanh(u)) == x * sigmoid(2u)
  const float c0 = 0.7978845608028654f;   // sqrt(2/pi)
  const float c1 = 0.044715f;
  float u = c0 * (x + c1 * x * x * x);
  float e = __expf(-2.0f * u);
  return x * __builtin_amdgcn_rcpf(1.0f + e);
}

// exactly one v_cvt_pk_bf16_f32 (lo -> [15:0], hi -> [31:16])
__device__ __forceinline__ unsigned pack2_bf16(float lo, float hi) {
  unsigned u;
  asm("v_cvt_pk_bf16_f32 %0, %1, %2" : "=v"(u) : "v"(lo), "v"(hi));
  return u;
}

// async DMA: 16 bytes global -> LDS, tracked by ASYNCcnt
__device__ __forceinline__ void async_copy_b128(unsigned lds_addr, const void* gptr) {
  unsigned long long ga = (unsigned long long)gptr;
  asm volatile("global_load_async_to_lds_b128 %0, %1, off"
               :: "v"(lds_addr), "v"(ga) : "memory");
}
__device__ __forceinline__ void wait_async0() {
  asm volatile("s_wait_asynccnt 0" ::: "memory");
}

// ---------------------------------------------------------------------------
// 0) zero output + per-expert counters
// ---------------------------------------------------------------------------
__global__ void zero_kernel(float* __restrict__ out, int n, int* __restrict__ counts) {
  for (int i = blockIdx.x * blockDim.x + threadIdx.x; i < n;
       i += gridDim.x * blockDim.x)
    out[i] = 0.0f;
  if (blockIdx.x == 0 && threadIdx.x < NEXP) counts[threadIdx.x] = 0;
}

// ---------------------------------------------------------------------------
// 1) hidden f32 -> bf16 prepass (A operand; 8 MB, stays L2-resident)
// ---------------------------------------------------------------------------
__global__ void cvt_hidden_kernel(const float* __restrict__ src,
                                  unsigned* __restrict__ dst) {
  int i = blockIdx.x * blockDim.x + threadIdx.x;   // one float4 -> uint2
  if (i >= TOK * DDIM / 4) return;
  float4 v = ((const float4*)src)[i];
  uint2 o;
  o.x = pack2_bf16(v.x, v.y);
  o.y = pack2_bf16(v.z, v.w);
  ((uint2*)dst)[i] = o;
}

// ---------------------------------------------------------------------------
// 2) weight transpose+convert prepass:
//    in  [E][KD][ND] f32  ->  out [E][ND][KD] bf16 (k-pairs packed per dword)
//    Output rows are exactly the bytes each GEMM LDS tile row needs, so the
//    GEMM can DMA them with global_load_async_to_lds_b128 (no VALU staging).
// ---------------------------------------------------------------------------
__global__ __launch_bounds__(256) void transpose_cvt_kernel(
    const float* __restrict__ in, unsigned* __restrict__ outdw,
    int KD, int ND) {
  __shared__ float tile[32][33];
  const int e   = blockIdx.z;
  const int kb  = blockIdx.y * 32;
  const int nbb = blockIdx.x * 32;
  const float* src = in + ((size_t)e * KD + kb) * ND + nbb;
#pragma unroll
  for (int p = 0; p < 4; ++p) {
    int idx = threadIdx.x + p * 256;
    int r = idx >> 5, c = idx & 31;
    tile[r][c] = src[(size_t)r * ND + c];
  }
  __syncthreads();
#pragma unroll
  for (int p = 0; p < 2; ++p) {
    int idx = threadIdx.x + p * 256;
    int n = idx >> 4, kd = idx & 15;
    unsigned v = pack2_bf16(tile[2 * kd][n], tile[2 * kd + 1][n]);
    outdw[(((size_t)e * ND + nbb + n) * KD + kb) / 2 + kd] = v;
  }
}

// ---------------------------------------------------------------------------
// 3) router observation stats: mean entropy + mean top-2 margin
// ---------------------------------------------------------------------------
__global__ void stats_kernel(const float* __restrict__ logits,
                             float* __restrict__ out_ent,
                             float* __restrict__ out_margin) {
  int t = blockIdx.x * blockDim.x + threadIdx.x;
  if (t >= TOK) return;
  const float* l = logits + (size_t)t * NEXP;
  float m = l[0];
#pragma unroll
  for (int i = 1; i < NEXP; ++i) m = fmaxf(m, l[i]);
  float s = 0.0f, p[NEXP];
#pragma unroll
  for (int i = 0; i < NEXP; ++i) { p[i] = __expf(l[i] - m); s += p[i]; }
  float inv = 1.0f / s;
  float ent = 0.0f, mx1 = -1.0f, mx2 = -1.0f;
#pragma unroll
  for (int i = 0; i < NEXP; ++i) {
    float pi = p[i] * inv;
    float lp = __logf(fmaxf(pi, 1e-9f));
    ent -= pi * lp;
    if (pi > mx1)      { mx2 = mx1; mx1 = pi; }
    else if (pi > mx2) { mx2 = pi; }
  }
  atomicAdd(out_ent,    ent * (1.0f / TOK));
  atomicAdd(out_margin, (mx1 - mx2) * (1.0f / TOK));
}

// ---------------------------------------------------------------------------
// 4/5/6) routing: count -> scan -> place (compact per-expert slot lists)
// ---------------------------------------------------------------------------
__global__ void count_kernel(const int* __restrict__ eidx, int* __restrict__ counts) {
  int t = blockIdx.x * blockDim.x + threadIdx.x;
  if (t < NPAIR) atomicAdd(&counts[eidx[t]], 1);
}

__global__ void scan_kernel(int* __restrict__ counts, int* __restrict__ offsets) {
  offsets[0] = 0;
  for (int e = 0; e < NEXP; ++e) {
    offsets[e + 1] = offsets[e] + counts[e];
    counts[e] = 0;  // reuse as placement cursor
  }
}

__global__ void place_kernel(const int* __restrict__ eidx,
                             const float* __restrict__ ewt,
                             const int* __restrict__ offsets,
                             int* __restrict__ cursors,
                             int* __restrict__ slot_token,
                             float* __restrict__ slot_weight) {
  int t = blockIdx.x * blockDim.x + threadIdx.x;
  if (t >= NPAIR) return;
  int e = eidx[t];
  int pos = offsets[e] + atomicAdd(&cursors[e], 1);
  slot_token[pos]  = t / TOPK;   // token index in [0, TOK)
  slot_weight[pos] = ewt[t];
}

// ---------------------------------------------------------------------------
// GEMM building blocks
//   LDS w-tile: [col 0..127][k-pair 0..15], row padded to 20 dwords (80 B:
//   keeps async B128 targets 16-B aligned, rows spread across banks)
// ---------------------------------------------------------------------------
__device__ __forceinline__ v16bf load_a_bf16(const unsigned short* ap) {
  v16bf a;
  uint4 u0 = *(const uint4*)(ap);        // 8 bf16: K = halfBase + [0,8)
  uint4 u1 = *(const uint4*)(ap + 8);    // 8 bf16: K = halfBase + [8,16)
  __builtin_memcpy(&a, &u0, 16);
  __builtin_memcpy(((char*)&a) + 16, &u1, 16);
  return a;
}

__device__ __forceinline__ v16bf load_frag(const unsigned (*buf)[20], int row, int klo) {
  const __bf16* rowp = (const __bf16*)&buf[row][0];
  v16bf b;
#pragma unroll
  for (int i = 0; i < 16; ++i) b[i] = rowp[klo + i];
  return b;
}

// ---------------------------------------------------------------------------
// 7) FFN1: h1 = gelu(X_group @ w1[e] + b1[e]) -> bf16 workspace
//    8 waves; tile M=128 x N=128; double-buffered LDS w-tile staged by
//    global_load_async_to_lds_b128; 8 WMMAs per wave per K-step
// ---------------------------------------------------------------------------
__global__ __launch_bounds__(256) void ffn1_kernel(
    const unsigned short* __restrict__ hidden_bf,
    const unsigned short* __restrict__ w1t,   // [E][F][D] bf16
    const float* __restrict__ b1,
    const int*   __restrict__ offsets,
    const int*   __restrict__ slot_token,
    unsigned short* __restrict__ h1out) {
  const int e    = blockIdx.z;
  const int base = offsets[e];
  const int cnt  = offsets[e + 1] - base;
  const int m0   = blockIdx.x * 128;
  if (m0 >= cnt) return;

  const int wave = threadIdx.x >> 5;   // 0..7 -> M-subtile
  const int lane = threadIdx.x & 31;
  const int mrow = lane & 15;
  const int half = lane >> 4;
  const int nb   = blockIdx.y * 128;
  const int klo  = half * 16;

  __shared__ unsigned ldsb[2][128][20];   // double-buffered w-tile (20 KB)

  // async staging: this thread owns column `col`, 16-B chunks {chk, chk+2}
  const int col = threadIdx.x & 127;
  const int chk = threadIdx.x >> 7;     // 0 or 1
  const char* wrow = (const char*)(w1t + ((size_t)e * FDIM + nb + col) * DDIM);
  const unsigned l0 = (unsigned)(size_t)&ldsb[0][col][0];
  const unsigned l1 = (unsigned)(size_t)&ldsb[1][col][0];

  int slotA  = m0 + wave * 16 + mrow;
  int tokenA = slot_token[base + ((slotA < cnt) ? slotA : 0)];
  const unsigned short* arow = hidden_bf + (size_t)tokenA * DDIM;

  const int KT = DDIM / 32;
  v8f acc[8] = {{}, {}, {}, {}, {}, {}, {}, {}};

  // prologue: DMA tile 0 into buffer 0 (each tile row = 64 B of wrow)
  async_copy_b128(l0 + chk * 16,       wrow + chk * 16);
  async_copy_b128(l0 + (chk + 2) * 16, wrow + (chk + 2) * 16);

  for (int kt = 0; kt < KT; ++kt) {
    wait_async0();        // our async copies (tile kt) complete
    __syncthreads();      // everyone's copies visible
    const int cur = kt & 1;
    // DMA next tile into the other buffer (overlaps the WMMAs below)
    if (kt + 1 < KT) {
      const char* g = wrow + (size_t)(kt + 1) * 64;
      unsigned ln = cur ? l0 : l1;
      async_copy_b128(ln + chk * 16,       g + chk * 16);
      async_copy_b128(ln + (chk + 2) * 16, g + (chk + 2) * 16);
    }
    if (kt + 2 < KT)
      __builtin_prefetch(wrow + (size_t)(kt + 2) * 64, 0, 3);

    v16bf a = load_a_bf16(arow + kt * 32 + half * 16);

    v16bf bcur = load_frag(ldsb[cur], mrow, klo);
#pragma unroll
    for (int j = 0; j < 8; ++j) {
      v16bf bnext;
      if (j < 7) bnext = load_frag(ldsb[cur], (j + 1) * 16 + mrow, klo);
      acc[j] = __builtin_amdgcn_wmma_f32_16x16x32_bf16(false, a, false, bcur,
                                                       (short)0, acc[j], false, false);
      bcur = bnext;
    }
  }

  // epilogue: bias + gelu -> bf16 activations
  float biasv[8];
#pragma unroll
  for (int j = 0; j < 8; ++j) biasv[j] = b1[e * FDIM + nb + j * 16 + mrow];
  const int mbase = m0 + wave * 16 + half * 8;
#pragma unroll
  for (int r = 0; r < 8; ++r) {
    int slot = mbase + r;
    if (slot < cnt) {
      unsigned short* orow = h1out + (size_t)(base + slot) * FDIM + nb + mrow;
#pragma unroll
      for (int j = 0; j < 8; ++j) {
        float v = gelu_fast(acc[j][r] + biasv[j]);
        __bf16 bv = (__bf16)v;
        unsigned short u;
        __builtin_memcpy(&u, &bv, 2);
        orow[j * 16] = u;
      }
    }
  }
}

// ---------------------------------------------------------------------------
// 8) FFN2: out += weight * (h1_group @ w2[e] + b2[e]) via atomic f32 scatter
// ---------------------------------------------------------------------------
__global__ __launch_bounds__(256) void ffn2_kernel(
    const unsigned short* __restrict__ h1,
    const unsigned short* __restrict__ w2t,   // [E][D][F] bf16
    const float* __restrict__ b2,
    const int*   __restrict__ offsets,
    const int*   __restrict__ slot_token,
    const float* __restrict__ slot_weight,
    float* __restrict__ out) {
  const int e    = blockIdx.z;
  const int base = offsets[e];
  const int cnt  = offsets[e + 1] - base;
  const int m0   = blockIdx.x * 128;
  if (m0 >= cnt) return;

  const int wave = threadIdx.x >> 5;
  const int lane = threadIdx.x & 31;
  const int mrow = lane & 15;
  const int half = lane >> 4;
  const int nb   = blockIdx.y * 128;
  const int klo  = half * 16;

  __shared__ unsigned ldsb[2][128][20];

  const int col = threadIdx.x & 127;
  const int chk = threadIdx.x >> 7;
  const char* wrow = (const char*)(w2t + ((size_t)e * DDIM + nb + col) * FDIM);
  const unsigned l0 = (unsigned)(size_t)&ldsb[0][col][0];
  const unsigned l1 = (unsigned)(size_t)&ldsb[1][col][0];

  int slotA = m0 + wave * 16 + mrow;
  const unsigned short* arow =
      h1 + (size_t)(base + ((slotA < cnt) ? slotA : 0)) * FDIM;

  const int KT = FDIM / 32;
  v8f acc[8] = {{}, {}, {}, {}, {}, {}, {}, {}};

  async_copy_b128(l0 + chk * 16,       wrow + chk * 16);
  async_copy_b128(l0 + (chk + 2) * 16, wrow + (chk + 2) * 16);

  for (int kt = 0; kt < KT; ++kt) {
    wait_async0();
    __syncthreads();
    const int cur = kt & 1;
    if (kt + 1 < KT) {
      const char* g = wrow + (size_t)(kt + 1) * 64;
      unsigned ln = cur ? l0 : l1;
      async_copy_b128(ln + chk * 16,       g + chk * 16);
      async_copy_b128(ln + (chk + 2) * 16, g + (chk + 2) * 16);
    }
    if (kt + 2 < KT)
      __builtin_prefetch(wrow + (size_t)(kt + 2) * 64, 0, 3);

    v16bf a = load_a_bf16(arow + kt * 32 + half * 16);

    v16bf bcur = load_frag(ldsb[cur], mrow, klo);
#pragma unroll
    for (int j = 0; j < 8; ++j) {
      v16bf bnext;
      if (j < 7) bnext = load_frag(ldsb[cur], (j + 1) * 16 + mrow, klo);
      acc[j] = __builtin_amdgcn_wmma_f32_16x16x32_bf16(false, a, false, bcur,
                                                       (short)0, acc[j], false, false);
      bcur = bnext;
    }
  }

  float biasv[8];
#pragma unroll
  for (int j = 0; j < 8; ++j) biasv[j] = b2[e * DDIM + nb + j * 16 + mrow];
  const int mbase = m0 + wave * 16 + half * 8;
#pragma unroll
  for (int r = 0; r < 8; ++r) {
    int slot = mbase + r;
    if (slot < cnt) {
      int   token = slot_token[base + slot];
      float wgt   = slot_weight[base + slot];
      float* orow = out + (size_t)token * DDIM + nb + mrow;
#pragma unroll
      for (int j = 0; j < 8; ++j)
        atomicAdd(orow + j * 16, (acc[j][r] + biasv[j]) * wgt);
    }
  }
}

// ---------------------------------------------------------------------------
// launcher
// ---------------------------------------------------------------------------
extern "C" void kernel_launch(void* const* d_in, const int* in_sizes, int n_in,
                              void* d_out, int out_size, void* d_ws, size_t ws_size,
                              hipStream_t stream) {
  const float* hidden = (const float*)d_in[0];
  const int*   eidx   = (const int*)d_in[1];
  const float* ewt    = (const float*)d_in[2];
  const float* rlog   = (const float*)d_in[3];
  const float* w1     = (const float*)d_in[4];
  const float* b1     = (const float*)d_in[5];
  const float* w2     = (const float*)d_in[6];
  const float* b2     = (const float*)d_in[7];
  float* out = (float*)d_out;

  // workspace layout
  char* ws = (char*)d_ws;
  int*   counts      = (int*)ws;                       // 8 ints (also cursors)
  int*   offsets     = counts + 8;                     // 9 ints
  int*   slot_token  = (int*)(ws + 256);               // NPAIR ints
  float* slot_weight = (float*)(ws + 256 + NPAIR * 4); // NPAIR floats
  size_t off = 131072;
  unsigned short* hid_bf = (unsigned short*)(ws + off); off += (size_t)TOK * DDIM * 2;     // 8 MB
  unsigned short* h1     = (unsigned short*)(ws + off); off += (size_t)NPAIR * FDIM * 2;   // 67 MB
  unsigned short* w1t    = (unsigned short*)(ws + off); off += (size_t)NEXP * DDIM * FDIM * 2; // 67 MB
  unsigned short* w2t    = (unsigned short*)(ws + off);                                    // 67 MB

  const int mixed_n = TOK * DDIM;          // 4,194,304
  float* out_ent    = out + mixed_n;
  float* out_margin = out + mixed_n + 1;

  // 0) zero mixed output + scalars + counters
  zero_kernel<<<256, 256, 0, stream>>>(out, out_size, counts);
  // 1) hidden -> bf16
  cvt_hidden_kernel<<<(TOK * DDIM / 4 + 255) / 256, 256, 0, stream>>>(
      hidden, (unsigned*)hid_bf);
  // 2) weight transpose+convert: w1 [E][D][F] -> w1t [E][F][D]; w2 likewise
  {
    dim3 g1(FDIM / 32, DDIM / 32, NEXP);   // (128, 32, 8)
    transpose_cvt_kernel<<<g1, 256, 0, stream>>>(w1, (unsigned*)w1t, DDIM, FDIM);
    dim3 g2(DDIM / 32, FDIM / 32, NEXP);   // (32, 128, 8)
    transpose_cvt_kernel<<<g2, 256, 0, stream>>>(w2, (unsigned*)w2t, FDIM, DDIM);
  }
  // 3) router stats
  stats_kernel<<<TOK / 256, 256, 0, stream>>>(rlog, out_ent, out_margin);
  // 4-6) routing
  count_kernel<<<NPAIR / 256, 256, 0, stream>>>(eidx, counts);
  scan_kernel<<<1, 1, 0, stream>>>(counts, offsets);
  place_kernel<<<NPAIR / 256, 256, 0, stream>>>(eidx, ewt, offsets, counts,
                                                slot_token, slot_weight);
  // 7) FFN1
  {
    dim3 grid(NPAIR / 128, FDIM / 128, NEXP);  // (64, 32, 8)
    ffn1_kernel<<<grid, 256, 0, stream>>>(hid_bf, w1t, b1, offsets, slot_token, h1);
  }
  // 8) FFN2 + weighted scatter
  {
    dim3 grid(NPAIR / 128, DDIM / 128, NEXP);  // (64, 8, 8)
    ffn2_kernel<<<grid, 256, 0, stream>>>(h1, w2t, b2, offsets, slot_token,
                                          slot_weight, out);
  }
}
